// MultiplicativeAttention_58145267254165
// MI455X (gfx1250) — matrix operand order repaired and checked by reading, not compiled, not run on hardware
//
#include <hip/hip_runtime.h>
#include <hip/hip_bf16.h>

// ---------------------------------------------------------------------------
// MultiplicativeAttention for MI455X (gfx1250, wave32, bf16 WMMA + async-LDS)
// ---------------------------------------------------------------------------

typedef __bf16 bf16_t;
typedef __attribute__((ext_vector_type(16))) __bf16 v16bf;
typedef __attribute__((ext_vector_type(8)))  float  v8f;
typedef __attribute__((ext_vector_type(4)))  int    v4i;

static constexpr int BATCH = 4;
static constexpr int SEQ   = 2048;
static constexpr int DMODEL= 1024;
static constexpr int NHEAD = 16;
static constexpr int HDIM  = 64;
static constexpr float INV_SCALE = 0.125f;       // 1/sqrt(64)
static constexpr float INV_H     = 1.0f / 16.0f;

static constexpr size_t NTOK   = (size_t)BATCH * SEQ;      // 8192
static constexpr size_t XELEMS = NTOK * DMODEL;            // 8,388,608
static constexpr size_t WELEMS = (size_t)DMODEL * DMODEL;  // 1,048,576

// ---- CDNA5 async global->LDS path (guarded; falls back to sync staging) ----
#if defined(__HIP_DEVICE_COMPILE__) && defined(__has_builtin)
#if __has_builtin(__builtin_amdgcn_global_load_async_to_lds_b128)
#define USE_ASYNC_LDS 1
#endif
#endif
#ifndef USE_ASYNC_LDS
#define USE_ASYNC_LDS 0
#endif

#if USE_ASYNC_LDS
typedef __attribute__((address_space(1))) v4i gv4i_t;  // global int4
typedef __attribute__((address_space(3))) v4i lv4i_t;  // LDS int4

__device__ __forceinline__ void async_copy_b128(const bf16_t* gsrc, bf16_t* ldst) {
  // generic LDS pointer low 32 bits == LDS byte offset (ISA 10.2 aperture rules)
  gv4i_t* g = (gv4i_t*)(unsigned long long)gsrc;
  lv4i_t* l = (lv4i_t*)(unsigned int)(unsigned long long)ldst;
  __builtin_amdgcn_global_load_async_to_lds_b128(g, l, 0, 0);
}
__device__ __forceinline__ void wait_async() {
#if __has_builtin(__builtin_amdgcn_s_wait_asynccnt)
  __builtin_amdgcn_s_wait_asynccnt(0);
#else
  asm volatile("s_wait_asynccnt 0x0" ::: "memory");
#endif
}
#else
__device__ __forceinline__ void wait_async() {}
#endif

__device__ __forceinline__ v8f wmma_bf16(v16bf a, v16bf b, v8f c) {
  return __builtin_amdgcn_wmma_f32_16x16x32_bf16(
      /*neg_a=*/false, a, /*neg_b=*/false, b,
      /*c_mod=*/(short)0, c, /*reuse_a=*/false, /*reuse_b=*/false);
}

// A fragment: 16x32 (MxK), row-major source with leading dim ld (elements).
// lanes 0-15 -> K {0..15}; lanes 16-31 -> K {16..31}.
__device__ __forceinline__ v16bf load_frag_a(const bf16_t* base, int ld) {
  const int lane = threadIdx.x & 31;
  const int m = lane & 15;
  const int koff = (lane & 16) ? 16 : 0;
  v16bf f;
#pragma unroll
  for (int p = 0; p < 8; ++p) {
    const int k = 2 * p + koff;
    f[2 * p]     = base[m * ld + k];
    f[2 * p + 1] = base[m * ld + k + 1];
  }
  return f;
}

// B fragment: 32x16 (KxN) where B[k][n] = src[n*ld + k] (source rows are N).
__device__ __forceinline__ v16bf load_frag_b_t(const bf16_t* base, int ld) {
  const int lane = threadIdx.x & 31;
  const int n = lane & 15;
  const int khalf = (lane & 16) ? 16 : 0;
  v16bf f;
#pragma unroll
  for (int p = 0; p < 8; ++p) {
    const int k = khalf + 2 * p;
    f[2 * p]     = base[n * ld + k];
    f[2 * p + 1] = base[n * ld + k + 1];
  }
  return f;
}

// B fragment: 32x16 (KxN) where B[k][n] = src[k*ld + n] (natural row-major).
__device__ __forceinline__ v16bf load_frag_b(const bf16_t* base, int ld) {
  const int lane = threadIdx.x & 31;
  const int n = lane & 15;
  const int khalf = (lane & 16) ? 16 : 0;
  v16bf f;
#pragma unroll
  for (int p = 0; p < 8; ++p) {
    const int k = khalf + 2 * p;
    f[2 * p]     = base[k * ld + n];
    f[2 * p + 1] = base[(k + 1) * ld + n];
  }
  return f;
}

// ---------------------------------------------------------------------------
// 1) f32 -> bf16 elementwise convert
// ---------------------------------------------------------------------------
__global__ void cvt_bf16_kernel(const float* __restrict__ src,
                                bf16_t* __restrict__ dst, int n) {
  const int i = blockIdx.x * blockDim.x + threadIdx.x;
  if (i < n) dst[i] = (bf16_t)src[i];
}

// ---------------------------------------------------------------------------
// 2) Fold A into Wk:  Wk_eff[h*64+e, c] = sum_d A[e,d] * Wk[h*64+d, c]
// ---------------------------------------------------------------------------
__global__ void fold_wk_kernel(const float* __restrict__ Wk,
                               const float* __restrict__ A,
                               const float* __restrict__ bk,
                               bf16_t* __restrict__ WkEff,
                               float* __restrict__ bkEff) {
  const int idx = blockIdx.x * blockDim.x + threadIdx.x;  // 0 .. 1M-1
  const int row = idx >> 10;
  const int col = idx & 1023;
  const int h = row >> 6;
  const int e = row & 63;
  float acc = 0.f;
#pragma unroll 8
  for (int d = 0; d < 64; ++d)
    acc += A[e * 64 + d] * Wk[(h * 64 + d) * 1024 + col];
  WkEff[row * 1024 + col] = (bf16_t)acc;
  if (col == 0) {
    float bacc = 0.f;
#pragma unroll 8
    for (int d = 0; d < 64; ++d) bacc += A[e * 64 + d] * bk[h * 64 + d];
    bkEff[row] = bacc;
  }
}

// ---------------------------------------------------------------------------
// 3/4) GEMM: Out = X(MxK,bf16) @ W^T(KxN,bf16) + bias
//    Block tile 128x64, 8 waves 4(M)x2(N), each wave 2x2 of 16x16 WMMA.
//    Double-buffered LDS; async global->LDS staging when available.
//    OUT_F32=0: bf16 out scattered to [B,H,S,64]; OUT_F32=1: f32 row-major.
// ---------------------------------------------------------------------------
#define LDT 40  // padded leading dim for LDS tiles (bank-spread, keeps 16B align)

template <int OUT_F32>
__global__ __launch_bounds__(256) void gemm_kernel(
    const bf16_t* __restrict__ X, const bf16_t* __restrict__ W,
    const float* __restrict__ bias, void* __restrict__ OutV) {
  __shared__ bf16_t As[2][128][LDT];
  __shared__ bf16_t Ws[2][64][LDT];

  const int tid = threadIdx.x;
  const int lane = tid & 31;
  const int wid = tid >> 5;
  const int wm = wid >> 1;   // 0..3
  const int wn = wid & 1;    // 0..1
  const int m_block = blockIdx.x * 128;
  const int n_block = blockIdx.y * 64;

  // per-thread staging coordinates (A: 128x32 -> 2x b128; W: 64x32 -> 1x b128)
  const int arow = tid >> 1, acol = (tid & 1) * 16;
  const int wrow = tid >> 2, wcol = (tid & 3) * 8;
  const bf16_t* gA = X + (size_t)(m_block + arow) * DMODEL + acol;
  const bf16_t* gW = W + (size_t)(n_block + wrow) * DMODEL + wcol;

  v8f acc[2][2];
#pragma unroll
  for (int i = 0; i < 2; ++i)
#pragma unroll
    for (int j = 0; j < 2; ++j) acc[i][j] = {};

  auto stage = [&](int buf, int k0) {
#if USE_ASYNC_LDS
    async_copy_b128(gA + k0,     &As[buf][arow][acol]);
    async_copy_b128(gA + k0 + 8, &As[buf][arow][acol + 8]);
    async_copy_b128(gW + k0,     &Ws[buf][wrow][wcol]);
#else
    *(uint4*)&As[buf][arow][acol]     = *(const uint4*)(gA + k0);
    *(uint4*)&As[buf][arow][acol + 8] = *(const uint4*)(gA + k0 + 8);
    *(uint4*)&Ws[buf][wrow][wcol]     = *(const uint4*)(gW + k0);
#endif
  };

  stage(0, 0);

  for (int k0 = 0, it = 0; k0 < DMODEL; k0 += 32, ++it) {
    const int cur = it & 1;
    wait_async();          // my tile-in-flight is complete
    __syncthreads();       // everyone's tile is visible; prev readers done
    if (k0 + 32 < DMODEL) stage(cur ^ 1, k0 + 32);  // overlap next copy w/ WMMA

    v16bf a0 = load_frag_a(&As[cur][wm * 32][0], LDT);
    v16bf a1 = load_frag_a(&As[cur][wm * 32 + 16][0], LDT);
    v16bf b0 = load_frag_b_t(&Ws[cur][wn * 32][0], LDT);
    v16bf b1 = load_frag_b_t(&Ws[cur][wn * 32 + 16][0], LDT);
    acc[0][0] = wmma_bf16(a0, b0, acc[0][0]);
    acc[0][1] = wmma_bf16(a0, b1, acc[0][1]);
    acc[1][0] = wmma_bf16(a1, b0, acc[1][0]);
    acc[1][1] = wmma_bf16(a1, b1, acc[1][1]);
  }

  // Epilogue
#pragma unroll
  for (int i = 0; i < 2; ++i) {
#pragma unroll
    for (int j = 0; j < 2; ++j) {
      const int n_g = n_block + wn * 32 + j * 16 + (lane & 15);
      const float bv = bias[n_g];
      const int h = n_g >> 6;
      const int d = n_g & 63;
#pragma unroll
      for (int r = 0; r < 8; ++r) {
        const int m_g = m_block + wm * 32 + i * 16 + ((lane >> 4) << 3) + r;
        if (OUT_F32) {
          ((float*)OutV)[(size_t)m_g * DMODEL + n_g] = acc[i][j][r] + bv;
        } else {
          const int b = m_g >> 11;
          const int s = m_g & 2047;
          ((bf16_t*)OutV)[(((size_t)b * NHEAD + h) * SEQ + s) * HDIM + d] =
              (bf16_t)(acc[i][j][r] + bv);
        }
      }
    }
  }
}

// ---------------------------------------------------------------------------
// 5) Flash attention: per wave one 16-row Q tile; online softmax over Sk.
// ---------------------------------------------------------------------------
__global__ __launch_bounds__(256) void flash_attn_kernel(
    const bf16_t* __restrict__ Qh, const bf16_t* __restrict__ Kh,
    const bf16_t* __restrict__ Vh, bf16_t* __restrict__ att,
    float* __restrict__ m_ws, float* __restrict__ l_ws) {
  __shared__ bf16_t Pst[8][16][32];   // per-wave P staging (D-frag -> A-frag)

  const int tid = threadIdx.x;
  const int lane = tid & 31;
  const int wv = tid >> 5;
  const int qt = blockIdx.x & 15;
  const int h  = (blockIdx.x >> 4) & 15;
  const int b  = blockIdx.x >> 8;
  const int q0 = qt * 128 + wv * 16;

  const size_t bh = (size_t)b * NHEAD + h;
  const bf16_t* Qb = Qh + (bh * SEQ + q0) * HDIM;
  const v16bf qa0 = load_frag_a(Qb, HDIM);        // d = 0..31
  const v16bf qa1 = load_frag_a(Qb + 32, HDIM);   // d = 32..63

  float mrow[8], lrow[8];
  v8f o[4];
#pragma unroll
  for (int r = 0; r < 8; ++r) { mrow[r] = -1.0e30f; lrow[r] = 0.f; }
#pragma unroll
  for (int d = 0; d < 4; ++d) o[d] = {};

  for (int k0 = 0; k0 < SEQ; k0 += 32) {
    const bf16_t* Kb = Kh + (bh * SEQ + k0) * HDIM;
    const bf16_t* Vb = Vh + (bh * SEQ + k0) * HDIM;

    // prefetch next K/V tiles (one 128B line per lane) -> global_prefetch_b8
    if (k0 + 32 < SEQ) {
      __builtin_prefetch(Kb + 32 * HDIM + lane * 64, 0, 3);
      __builtin_prefetch(Vb + 32 * HDIM + lane * 64, 0, 3);
    }

    // scores tile 16x32 = two 16x16 D-frags, K-dim = head dim (2 halves each)
    v8f s0 = {}, s1 = {};
    s0 = wmma_bf16(qa0, load_frag_b_t(Kb, HDIM), s0);
    s0 = wmma_bf16(qa1, load_frag_b_t(Kb + 32, HDIM), s0);
    s1 = wmma_bf16(qa0, load_frag_b_t(Kb + 16 * HDIM, HDIM), s1);
    s1 = wmma_bf16(qa1, load_frag_b_t(Kb + 16 * HDIM + 32, HDIM), s1);

    float p0[8], p1[8], mnew[8], cor[8];
#pragma unroll
    for (int r = 0; r < 8; ++r) {
      p0[r] = s0[r] * INV_SCALE;
      p1[r] = s1[r] * INV_SCALE;
    }
    // row reductions across the 16 lanes that share a row (width-16 shuffles)
#pragma unroll
    for (int r = 0; r < 8; ++r) {
      float t = fmaxf(p0[r], p1[r]);
#pragma unroll
      for (int msk = 8; msk >= 1; msk >>= 1)
        t = fmaxf(t, __shfl_xor(t, msk, 16));
      mnew[r] = fmaxf(mrow[r], t);
      cor[r] = __expf(mrow[r] - mnew[r]);
      p0[r] = __expf(p0[r] - mnew[r]);
      p1[r] = __expf(p1[r] - mnew[r]);
      float rs = p0[r] + p1[r];
#pragma unroll
      for (int msk = 8; msk >= 1; msk >>= 1)
        rs += __shfl_xor(rs, msk, 16);
      lrow[r] = lrow[r] * cor[r] + rs;
      mrow[r] = mnew[r];
    }
#pragma unroll
    for (int d = 0; d < 4; ++d)
#pragma unroll
      for (int r = 0; r < 8; ++r) o[d][r] *= cor[r];

    // D-frag layout -> A-frag layout through LDS
    __syncthreads();
    {
      const int n = lane & 15;
      const int rbase = (lane >> 4) << 3;
#pragma unroll
      for (int r = 0; r < 8; ++r) {
        Pst[wv][rbase + r][n]      = (bf16_t)p0[r];
        Pst[wv][rbase + r][n + 16] = (bf16_t)p1[r];
      }
    }
    __syncthreads();
    const v16bf pa = load_frag_a(&Pst[wv][0][0], 32);

    // O += P @ V  (K = 32 key positions, 4 d-blocks of 16)
#pragma unroll
    for (int d = 0; d < 4; ++d)
      o[d] = wmma_bf16(pa, load_frag_b(Vb + d * 16, HDIM), o[d]);
  }

  // normalize, store attended as bf16 into [B, S, H*64] row-major
#pragma unroll
  for (int d = 0; d < 4; ++d) {
    const int n = lane & 15;
#pragma unroll
    for (int r = 0; r < 8; ++r) {
      const int row = q0 + ((lane >> 4) << 3) + r;
      const float val = o[d][r] / lrow[r];
      att[((size_t)b * SEQ + row) * DMODEL + h * HDIM + d * 16 + n] = (bf16_t)val;
    }
  }
  if ((lane & 15) == 0) {
#pragma unroll
    for (int r = 0; r < 8; ++r) {
      const size_t idx = bh * SEQ + q0 + ((lane >> 4) << 3) + r;
      m_ws[idx] = mrow[r];
      l_ws[idx] = lrow[r];
    }
  }
}

// ---------------------------------------------------------------------------
// 6) avg_weights: recompute score tiles per head with WMMA, normalize with
//    stored (m,l), average over heads. Each wave owns one 16x16 (q,k) tile.
// ---------------------------------------------------------------------------
__global__ __launch_bounds__(256) void avg_weights_kernel(
    const bf16_t* __restrict__ Qh, const bf16_t* __restrict__ Kh,
    const float* __restrict__ m_ws, const float* __restrict__ l_ws,
    float* __restrict__ avg) {
  const int tid = threadIdx.x;
  const int lane = tid & 31;
  const int wv = tid >> 5;
  const int k0 = blockIdx.x * 128 + wv * 16;
  const int q0 = blockIdx.y * 16;
  const int b  = blockIdx.z;

  v8f acc = {};
  for (int h = 0; h < NHEAD; ++h) {
    const size_t bh = (size_t)b * NHEAD + h;
    const bf16_t* Qb = Qh + (bh * SEQ + q0) * HDIM;
    const bf16_t* Kb = Kh + (bh * SEQ + k0) * HDIM;
    v8f s = {};
    s = wmma_bf16(load_frag_a(Qb, HDIM),      load_frag_b_t(Kb, HDIM),      s);
    s = wmma_bf16(load_frag_a(Qb + 32, HDIM), load_frag_b_t(Kb + 32, HDIM), s);
#pragma unroll
    for (int r = 0; r < 8; ++r) {
      const size_t idx = bh * SEQ + q0 + ((lane >> 4) << 3) + r;
      const float m = m_ws[idx];
      const float l = l_ws[idx];
      acc[r] += __expf(s[r] * INV_SCALE - m) * (INV_H / l);
    }
  }
#pragma unroll
  for (int r = 0; r < 8; ++r) {
    const int q = q0 + ((lane >> 4) << 3) + r;
    const int k = k0 + (lane & 15);
    avg[((size_t)b * SEQ + q) * SEQ + k] = acc[r];
  }
}

// ---------------------------------------------------------------------------
// Host side
// ---------------------------------------------------------------------------
extern "C" void kernel_launch(void* const* d_in, const int* in_sizes, int n_in,
                              void* d_out, int out_size, void* d_ws, size_t ws_size,
                              hipStream_t stream) {
  const float* q_in = (const float*)d_in[0];
  const float* k_in = (const float*)d_in[1];
  const float* v_in = (const float*)d_in[2];
  const float* Wq   = (const float*)d_in[3];
  const float* bq   = (const float*)d_in[4];
  const float* Wk   = (const float*)d_in[5];
  const float* bk   = (const float*)d_in[6];
  const float* Wv   = (const float*)d_in[7];
  const float* bv   = (const float*)d_in[8];
  const float* A    = (const float*)d_in[9];
  const float* Wo   = (const float*)d_in[10];
  const float* bo   = (const float*)d_in[11];

  char* ws = (char*)d_ws;
  size_t off = 0;
  auto take = [&](size_t bytes) { char* p = ws + off; off += (bytes + 255) & ~(size_t)255; return p; };

  bf16_t* qbf   = (bf16_t*)take(XELEMS * 2);
  bf16_t* kbf   = (bf16_t*)take(XELEMS * 2);
  bf16_t* vbf   = (bf16_t*)take(XELEMS * 2);
  bf16_t* wq_bf = (bf16_t*)take(WELEMS * 2);
  bf16_t* wke_bf= (bf16_t*)take(WELEMS * 2);
  bf16_t* wv_bf = (bf16_t*)take(WELEMS * 2);
  bf16_t* wo_bf = (bf16_t*)take(WELEMS * 2);
  float*  bke   = (float*)take(DMODEL * 4);
  bf16_t* Qh    = (bf16_t*)take(XELEMS * 2);
  bf16_t* Kwh   = (bf16_t*)take(XELEMS * 2);
  bf16_t* Vhh   = (bf16_t*)take(XELEMS * 2);
  bf16_t* att   = (bf16_t*)take(XELEMS * 2);
  float*  m_ws  = (float*)take((XELEMS / HDIM) * 4);
  float*  l_ws  = (float*)take((XELEMS / HDIM) * 4);

  float* out_main = (float*)d_out;                 // [B,S,1024]
  float* out_avg  = out_main + XELEMS;             // [B,S,S]

  const int T = 256;
  // 1) converts
  cvt_bf16_kernel<<<(int)(XELEMS / T), T, 0, stream>>>(q_in, qbf, (int)XELEMS);
  cvt_bf16_kernel<<<(int)(XELEMS / T), T, 0, stream>>>(k_in, kbf, (int)XELEMS);
  cvt_bf16_kernel<<<(int)(XELEMS / T), T, 0, stream>>>(v_in, vbf, (int)XELEMS);
  cvt_bf16_kernel<<<(int)(WELEMS / T), T, 0, stream>>>(Wq, wq_bf, (int)WELEMS);
  cvt_bf16_kernel<<<(int)(WELEMS / T), T, 0, stream>>>(Wv, wv_bf, (int)WELEMS);
  cvt_bf16_kernel<<<(int)(WELEMS / T), T, 0, stream>>>(Wo, wo_bf, (int)WELEMS);
  // 2) fold A into Wk
  fold_wk_kernel<<<(int)(WELEMS / T), T, 0, stream>>>(Wk, A, bk, wke_bf, bke);
  // 3) projections -> [B,H,S,64] bf16
  dim3 ggrid(NTOK / 128, DMODEL / 64);
  gemm_kernel<0><<<ggrid, T, 0, stream>>>(qbf, wq_bf, bq, Qh);
  gemm_kernel<0><<<ggrid, T, 0, stream>>>(kbf, wke_bf, bke, Kwh);
  gemm_kernel<0><<<ggrid, T, 0, stream>>>(vbf, wv_bf, bv, Vhh);
  // 4) flash attention
  flash_attn_kernel<<<BATCH * NHEAD * (SEQ / 128), T, 0, stream>>>(
      Qh, Kwh, Vhh, att, m_ws, l_ws);
  // 5) output projection (f32)
  gemm_kernel<1><<<ggrid, T, 0, stream>>>(att, wo_bf, bo, out_main);
  // 6) averaged attention weights
  dim3 agrid(SEQ / 128, SEQ / 16, BATCH);
  avg_weights_kernel<<<agrid, T, 0, stream>>>(Qh, Kwh, m_ws, l_ws, out_avg);

  (void)in_sizes; (void)n_in; (void)out_size; (void)ws_size;
}
